// MultiheadSelfAttention_55233279427185
// MI455X (gfx1250) — matrix operand order repaired and checked by reading, not compiled, not run on hardware
//
#include <hip/hip_runtime.h>
#include <hip/hip_bf16.h>

#define DIM    1024
#define RANK   64
#define NHEADS 16
#define BATCH  2
#define SEQ    2048

typedef __bf16 bf16;
typedef __attribute__((ext_vector_type(16))) __bf16 v16bf;
typedef __attribute__((ext_vector_type(8)))  __bf16 v8bf;
typedef __attribute__((ext_vector_type(8)))  float  v8f;
typedef __attribute__((ext_vector_type(4)))  int    v4i;

// ---------------------------------------------------------------------------
// CDNA5 async global->LDS path (ASYNCcnt), with portable fallbacks.
// Probe result: __builtin_amdgcn_global_load_async_to_lds_b128 exists and its
// first parameter is 'int __vector(4) *' (global AS) -> typed v4i pointers.
// ---------------------------------------------------------------------------
#if __has_builtin(__builtin_amdgcn_global_load_async_to_lds_b128)
#define HAVE_ASYNC_LDS 1
#endif

__device__ __forceinline__ void copy16_g2l(const bf16* g, bf16* l) {
#ifdef HAVE_ASYNC_LDS
  typedef __attribute__((address_space(1))) v4i* gp128;
  typedef __attribute__((address_space(3))) v4i* lp128;
  __builtin_amdgcn_global_load_async_to_lds_b128(
      (gp128)(unsigned long long)g,   // global/generic share addresses: inttoptr
      (lp128)l,                       // explicit addrspacecast generic->LDS
      0, 0);
#else
  *(v8bf*)l = *(const v8bf*)g;  // synchronous fallback (16B)
#endif
}

#if __has_builtin(__builtin_amdgcn_s_wait_asynccnt)
#define WAIT_ASYNC(n) __builtin_amdgcn_s_wait_asynccnt(n)
#elif defined(HAVE_ASYNC_LDS)
#define WAIT_ASYNC(n) asm volatile("s_wait_asynccnt %0" ::"i"(n) : "memory")
#else
#define WAIT_ASYNC(n)
#endif

// ---------------------------------------------------------------------------
// WMMA fragment helpers (bf16 16x16x32, wave32; layouts per CDNA5 ISA 7.12.2)
// ---------------------------------------------------------------------------
__device__ __forceinline__ v16bf frag_cat(v8bf lo, v8bf hi) {
  return __builtin_shufflevector(lo, hi, 0,1,2,3,4,5,6,7,8,9,10,11,12,13,14,15);
}

// A fragment: 16x32 bf16 from row-major [m][k] storage, row stride ld (elems).
__device__ __forceinline__ v16bf load_A(const bf16* base, int ld, int lane) {
  const int m = lane & 15, half = lane >> 4;
  const bf16* row = base + (size_t)m * ld;
  v8bf lo = *(const v8bf*)(row + 8 * half);
  v8bf hi = *(const v8bf*)(row + 16 + 8 * half);
  return frag_cat(lo, hi);
}

// B fragment: 32x16 bf16 from B^T row-major [n][k] storage, row stride ld (elems).
__device__ __forceinline__ v16bf load_B(const bf16* baseT, int ld, int lane) {
  const int n = lane & 15, half = lane >> 4;
  const bf16* row = baseT + (size_t)n * ld;
  v8bf lo = *(const v8bf*)(row + 8 * half);
  v8bf hi = *(const v8bf*)(row + 16 + 8 * half);
  return frag_cat(lo, hi);
}

__device__ __forceinline__ v8f wmma_bf16(v16bf a, v16bf b, v8f c) {
  return __builtin_amdgcn_wmma_f32_16x16x32_bf16(false, a, false, b, (short)0, c,
                                                 false, false);
}

// ---------------------------------------------------------------------------
// Prep kernels: fp32 -> bf16 conversions + transposes into WMMA-friendly layouts
// ---------------------------------------------------------------------------
__global__ void k_prep_x(const float* __restrict__ x, bf16* __restrict__ xb,
                         bf16* __restrict__ xT) {
  __shared__ float tile[32][33];
  const int b  = blockIdx.z;
  const int t0 = blockIdx.x * 32, d0 = blockIdx.y * 32;
  const int tx = threadIdx.x, ty = threadIdx.y;  // 32 x 8
  for (int i = 0; i < 32; i += 8) {
    float v = x[((size_t)b * SEQ + t0 + ty + i) * DIM + d0 + tx];
    xb[((size_t)b * SEQ + t0 + ty + i) * DIM + d0 + tx] = (bf16)v;
    tile[ty + i][tx] = v;
  }
  __syncthreads();
  for (int i = 0; i < 32; i += 8)
    xT[((size_t)b * DIM + d0 + ty + i) * SEQ + t0 + tx] = (bf16)tile[tx][ty + i];
}

__global__ void k_prep_qk(const float* __restrict__ Q, const float* __restrict__ K,
                          bf16* __restrict__ Qb, bf16* __restrict__ Kb) {
  const size_t i = (size_t)blockIdx.x * 256 + threadIdx.x;
  Qb[i] = (bf16)(Q[i] * 0.03125f);  // fold 1/sqrt(DIM), exact power of two
  Kb[i] = (bf16)K[i];
}

__global__ void k_prep_vo(const float* __restrict__ VO, bf16* __restrict__ VOt) {
  __shared__ float tile[32][33];
  const int h  = blockIdx.z;
  const int d0 = blockIdx.x * 32, e0 = blockIdx.y * 32;
  const int tx = threadIdx.x, ty = threadIdx.y;
  const float* VOh = VO + (size_t)h * DIM * DIM;
  bf16* VOth = VOt + (size_t)h * DIM * DIM;
  for (int i = 0; i < 32; i += 8)
    tile[ty + i][tx] = VOh[(size_t)(d0 + ty + i) * DIM + e0 + tx];
  __syncthreads();
  for (int i = 0; i < 32; i += 8)
    VOth[(size_t)(e0 + ty + i) * DIM + d0 + tx] = (bf16)tile[tx][ty + i];
}

// ---------------------------------------------------------------------------
// Stage 1: xq/xk[b][h][t][r] = X @ Q_h^T, X @ K_h^T  (bf16 WMMA)
// ---------------------------------------------------------------------------
__global__ void k_project(const bf16* __restrict__ xb, const bf16* __restrict__ Qb,
                          const bf16* __restrict__ Kb, bf16* __restrict__ xq,
                          bf16* __restrict__ xk) {
  const int lane = threadIdx.x & 31, wave = threadIdx.x >> 5;
  const int b = blockIdx.z, h = blockIdx.y;
  const int m0 = blockIdx.x * 128 + wave * 16;
  const bf16* Arow = xb + ((size_t)b * SEQ + m0) * DIM;
  const bf16* Qh = Qb + (size_t)h * RANK * DIM;
  const bf16* Kh = Kb + (size_t)h * RANK * DIM;
  const v8f vzero = {};
  v8f accq[4] = {vzero, vzero, vzero, vzero};
  v8f acck[4] = {vzero, vzero, vzero, vzero};
  for (int kk = 0; kk < DIM; kk += 32) {
    v16bf a = load_A(Arow + kk, DIM, lane);
#pragma unroll
    for (int nt = 0; nt < 4; ++nt) {
      v16bf bq = load_B(Qh + (size_t)nt * 16 * DIM + kk, DIM, lane);
      accq[nt] = wmma_bf16(a, bq, accq[nt]);
      v16bf bk = load_B(Kh + (size_t)nt * 16 * DIM + kk, DIM, lane);
      acck[nt] = wmma_bf16(a, bk, acck[nt]);
    }
  }
  const int n = lane & 15, half = lane >> 4;
  bf16* oq = xq + (((size_t)b * NHEADS + h) * SEQ + m0) * RANK;
  bf16* ok = xk + (((size_t)b * NHEADS + h) * SEQ + m0) * RANK;
#pragma unroll
  for (int nt = 0; nt < 4; ++nt)
#pragma unroll
    for (int v = 0; v < 8; ++v) {
      const int m = v + 8 * half;
      oq[(size_t)m * RANK + nt * 16 + n] = (bf16)accq[nt][v];
      ok[(size_t)m * RANK + nt * 16 + n] = (bf16)acck[nt][v];
    }
}

// ---------------------------------------------------------------------------
// Stage 2: flash attention with double-buffered ASYNC global->LDS staging of
// the X^T value tiles. grid(SEQ/32, NHEADS, BATCH), 8 waves:
//   rg = wave&1 (16-row group), dc = wave>>1 (256-wide d-chunk).
// Per 64-key step the WG stages [1024 d][64 u] bf16 into LDS (padded rows,
// 144B stride -> conflict-free ds_load_b128), prefetching the next step's
// tile while computing scores, then consumes it as WMMA B-fragments.
// ---------------------------------------------------------------------------
#define VROWPAD 72                       // 64 u + 8 pad elems (16B-aligned rows)
#define VCHUNK  (256 * VROWPAD)          // one 256-wide d-chunk
#define VBUF    (4 * VCHUNK)             // all four d-chunks
#define NSTEP   (SEQ / 64)

__global__ void k_attention(const bf16* __restrict__ xq, const bf16* __restrict__ xk,
                            const bf16* __restrict__ xT, bf16* __restrict__ Cb) {
  extern __shared__ __align__(16) char dyn_smem[];
  bf16* vbuf = (bf16*)dyn_smem;          // [2][4][256][VROWPAD]

  const int lane = threadIdx.x & 31, wave = threadIdx.x >> 5;
  const int rg = wave & 1, dc = wave >> 1;
  const int b = blockIdx.z, h = blockIdx.y;
  const int row0 = blockIdx.x * 32 + rg * 16;
  const int d0 = dc * 256;
  const int n = lane & 15, half = lane >> 4;

  __shared__ __align__(16) bf16 Plds[2][16][80];
  __shared__ float s_alpha[2][16];
  __shared__ float s_invl[2][16];

  const bf16* xqh = xq + (((size_t)b * NHEADS + h) * SEQ) * RANK;
  const bf16* xkh = xk + (((size_t)b * NHEADS + h) * SEQ) * RANK;

  // Staging geometry: this wave fills rows [rg*128, rg*128+128) of chunk dc.
  // Each async b128 moves 4 rows x 16B: lane -> (row += lane>>3, useg = lane&7).
  const int srow = rg * 128 + (lane >> 3);        // chunk-local row (+4 per instr)
  const int sseg = (lane & 7) * 8;                // u offset (elems)
  const bf16* gstage = xT + ((size_t)b * DIM + d0 + srow) * SEQ + sseg;
  bf16* lstage = vbuf + (dc * 256 + srow) * VROWPAD + sseg;

  auto stage = [&](int u0, int buf) {
    const bf16* g = gstage + u0;
    bf16* l = lstage + buf * VBUF;
#pragma unroll
    for (int i = 0; i < 32; ++i)
      copy16_g2l(g + (size_t)4 * i * SEQ, l + 4 * i * VROWPAD);
  };

  // Xq A-fragments for this row group: invariant over the key loop.
  const v16bf aq0 = load_A(xqh + (size_t)row0 * RANK + 0, RANK, lane);
  const v16bf aq1 = load_A(xqh + (size_t)row0 * RANK + 32, RANK, lane);

  const v8f vzero = {};
  v8f acc[16];
#pragma unroll
  for (int i = 0; i < 16; ++i) acc[i] = vzero;

  float m_run[8], l_run[8];
#pragma unroll
  for (int v = 0; v < 8; ++v) { m_run[v] = -INFINITY; l_run[v] = 0.0f; }

  stage(0, 0);  // prologue fill of buffer 0

  for (int step = 0; step < NSTEP; ++step) {
    const int u0 = step * 64;
    if (step + 1 < NSTEP) stage(u0 + 64, (step + 1) & 1);  // async prefetch

    if (dc == 0) {  // wave-uniform branch -> EXEC all-ones around WMMA
      v8f s[4];
#pragma unroll
      for (int nt = 0; nt < 4; ++nt) {
        v8f c = vzero;
        v16bf bk0 = load_B(xkh + (size_t)(u0 + nt * 16) * RANK + 0, RANK, lane);
        c = wmma_bf16(aq0, bk0, c);
        v16bf bk1 = load_B(xkh + (size_t)(u0 + nt * 16) * RANK + 32, RANK, lane);
        c = wmma_bf16(aq1, bk1, c);
        s[nt] = c;
      }
      // online softmax: row m = v + 8*half lives in a 16-lane half
#pragma unroll
      for (int v = 0; v < 8; ++v) {
        float mx = fmaxf(fmaxf(s[0][v], s[1][v]), fmaxf(s[2][v], s[3][v]));
#pragma unroll
        for (int off = 8; off >= 1; off >>= 1)
          mx = fmaxf(mx, __shfl_xor(mx, off, 16));
        const float mnew = fmaxf(m_run[v], mx);
        const float alpha = __expf(m_run[v] - mnew);
        m_run[v] = mnew;
        float rsum = 0.0f;
#pragma unroll
        for (int nt = 0; nt < 4; ++nt) {
          const float p = __expf(s[nt][v] - mnew);
          s[nt][v] = p;
          rsum += p;
        }
#pragma unroll
        for (int off = 8; off >= 1; off >>= 1)
          rsum += __shfl_xor(rsum, off, 16);
        l_run[v] = l_run[v] * alpha + rsum;
        if (n == 0) s_alpha[rg][v + 8 * half] = alpha;
      }
#pragma unroll
      for (int nt = 0; nt < 4; ++nt)
#pragma unroll
        for (int v = 0; v < 8; ++v)
          Plds[rg][v + 8 * half][nt * 16 + n] = (bf16)s[nt][v];
    }

    // Retire this step's 32 async fills (oldest; loads complete in order),
    // then make all waves' fills + P visible.
    if (step + 1 < NSTEP) { WAIT_ASYNC(32); } else { WAIT_ASYNC(0); }
    __syncthreads();

    // rescale accumulators, then P @ V from LDS for this wave's d-chunk
#pragma unroll
    for (int v = 0; v < 8; ++v) {
      const float alpha = s_alpha[rg][v + 8 * half];
#pragma unroll
      for (int nt = 0; nt < 16; ++nt) acc[nt][v] *= alpha;
    }
    const bf16* chunk = vbuf + (size_t)(step & 1) * VBUF + dc * VCHUNK;
#pragma unroll
    for (int ks = 0; ks < 2; ++ks) {
      v16bf ap = load_A(&Plds[rg][0][0] + ks * 32, 80, lane);
#pragma unroll
      for (int nt = 0; nt < 16; ++nt) {
        v16bf bv = load_B(chunk + (size_t)nt * 16 * VROWPAD + ks * 32, VROWPAD, lane);
        acc[nt] = wmma_bf16(ap, bv, acc[nt]);
      }
    }
    __syncthreads();  // protects P and buffer (step&1) reuse
  }

  if (dc == 0 && n == 0) {
#pragma unroll
    for (int v = 0; v < 8; ++v) s_invl[rg][v + 8 * half] = 1.0f / l_run[v];
  }
  __syncthreads();

  bf16* Crow = Cb + (((size_t)b * SEQ + row0) * NHEADS + h) * DIM + d0;
#pragma unroll
  for (int v = 0; v < 8; ++v) {
    const float inv = s_invl[rg][v + 8 * half];
    const int m = v + 8 * half;
#pragma unroll
    for (int nt = 0; nt < 16; ++nt)
      Crow[(size_t)m * NHEADS * DIM + nt * 16 + n] = (bf16)(acc[nt][v] * inv);
  }
}

// ---------------------------------------------------------------------------
// Stage 3: out = C[4096 x 16384] @ W[16384 x 1024], W[(h,d)][e] via VOt[h][e][d]
// ---------------------------------------------------------------------------
__global__ void k_out_gemm(const bf16* __restrict__ Cb, const bf16* __restrict__ VOt,
                           float* __restrict__ out) {
  const int lane = threadIdx.x & 31, wave = threadIdx.x >> 5;
  const int row0 = blockIdx.x * 128 + wave * 16;
  const int e0 = blockIdx.y * 64;
  const int KTOT = NHEADS * DIM;  // 16384
  const bf16* Arow = Cb + (size_t)row0 * KTOT;
  const v8f vzero = {};
  v8f acc[4] = {vzero, vzero, vzero, vzero};
  for (int ks = 0; ks < KTOT; ks += 32) {
    __builtin_prefetch(Arow + (size_t)(lane & 15) * KTOT + ks + 256, 0, 1);
    v16bf a = load_A(Arow + ks, KTOT, lane);
    const int h = ks >> 10;
    const int dk = ks & (DIM - 1);
    const bf16* Bt = VOt + ((size_t)h * DIM + e0) * DIM + dk;
#pragma unroll
    for (int nt = 0; nt < 4; ++nt) {
      v16bf bw = load_B(Bt + (size_t)nt * 16 * DIM, DIM, lane);
      acc[nt] = wmma_bf16(a, bw, acc[nt]);
    }
  }
  const int n = lane & 15, half = lane >> 4;
#pragma unroll
  for (int nt = 0; nt < 4; ++nt)
#pragma unroll
    for (int v = 0; v < 8; ++v)
      out[(size_t)(row0 + v + 8 * half) * DIM + e0 + nt * 16 + n] = acc[nt][v];
}

// ---------------------------------------------------------------------------
extern "C" void kernel_launch(void* const* d_in, const int* in_sizes, int n_in,
                              void* d_out, int out_size, void* d_ws, size_t ws_size,
                              hipStream_t stream) {
  const float* x  = (const float*)d_in[0];
  const float* Q  = (const float*)d_in[1];
  const float* K  = (const float*)d_in[2];
  const float* VO = (const float*)d_in[3];
  float* out = (float*)d_out;

  char* ws = (char*)d_ws;
  size_t off = 0;
  auto take = [&](size_t elems) {
    bf16* p = (bf16*)(ws + off);
    off += elems * sizeof(bf16);
    return p;
  };
  bf16* xb  = take((size_t)BATCH * SEQ * DIM);
  bf16* xT  = take((size_t)BATCH * DIM * SEQ);
  bf16* Qb  = take((size_t)NHEADS * RANK * DIM);
  bf16* Kb  = take((size_t)NHEADS * RANK * DIM);
  bf16* VOt = take((size_t)NHEADS * DIM * DIM);
  bf16* xq  = take((size_t)BATCH * NHEADS * SEQ * RANK);
  bf16* xk  = take((size_t)BATCH * NHEADS * SEQ * RANK);
  bf16* Cb  = take((size_t)BATCH * SEQ * NHEADS * DIM);

  const size_t attn_lds = (size_t)2 * VBUF * sizeof(bf16);  // 288 KB dynamic

  k_prep_x<<<dim3(SEQ / 32, DIM / 32, BATCH), dim3(32, 8), 0, stream>>>(x, xb, xT);
  k_prep_qk<<<(NHEADS * RANK * DIM) / 256, 256, 0, stream>>>(Q, K, Qb, Kb);
  k_prep_vo<<<dim3(DIM / 32, DIM / 32, NHEADS), dim3(32, 8), 0, stream>>>(VO, VOt);
  k_project<<<dim3(SEQ / 128, NHEADS, BATCH), 256, 0, stream>>>(xb, Qb, Kb, xq, xk);
  k_attention<<<dim3(SEQ / 32, NHEADS, BATCH), 256, attn_lds, stream>>>(xq, xk, xT, Cb);
  k_out_gemm<<<dim3((BATCH * SEQ) / 128, DIM / 64), 256, 0, stream>>>(Cb, VOt, out);
}
